// ViT_4037269258992
// MI455X (gfx1250) — matrix operand order repaired and checked by reading, not compiled, not run on hardware
//
#include <hip/hip_runtime.h>
#include <hip/hip_bf16.h>

typedef unsigned short ush;
typedef __attribute__((ext_vector_type(16))) __bf16 v16bf;
typedef __attribute__((ext_vector_type(2)))  __bf16 v2bf;
typedef __attribute__((ext_vector_type(8)))  float  v8f;

union Frag {
    v16bf v;
    ush   u[16];
    uint4 q[2];
};

__device__ __forceinline__ ush f2bf(float f) {
    unsigned int u = __float_as_uint(f);
    unsigned int r = u + 0x7FFFu + ((u >> 16) & 1u);   // round-to-nearest-even
    return (ush)(r >> 16);
}

// pack two floats -> packed bf16x2 (v_cvt_pk_bf16_f32 when available)
__device__ __forceinline__ unsigned pack2bf(float a, float b) {
#if defined(__has_builtin) && __has_builtin(__builtin_amdgcn_cvt_pk_bf16_f32)
    union { v2bf h; unsigned u; } cv;
    cv.h = __builtin_amdgcn_cvt_pk_bf16_f32(a, b);
    return cv.u;
#else
    return (unsigned)f2bf(a) | ((unsigned)f2bf(b) << 16);
#endif
}

__device__ __forceinline__ v8f wmma_bf16(const Frag& a, const Frag& b, v8f c) {
    return __builtin_amdgcn_wmma_f32_16x16x32_bf16(
        /*neg_a=*/false, a.v, /*neg_b=*/false, b.v,
        /*c_mod=*/(short)0, c, /*reuse_a=*/false, /*reuse_b=*/false);
}

#define ZERO8 {0.f,0.f,0.f,0.f,0.f,0.f,0.f,0.f}

// ---------------------------------------------------------------------------
// LayerNorm: one block per row of 1024, fp32 stats, bf16 output
// ---------------------------------------------------------------------------
__global__ __launch_bounds__(256)
void ln_kernel(const float* __restrict__ x, const float* __restrict__ g,
               const float* __restrict__ b, ush* __restrict__ out)
{
    __shared__ float s1[256], s2[256];
    const int row = blockIdx.x;
    const float* xr = x + (size_t)row * 1024;
    float v[4], sum = 0.f, sq = 0.f;
#pragma unroll
    for (int i = 0; i < 4; i++) {
        v[i] = xr[threadIdx.x + i * 256];
        sum += v[i];
        sq  += v[i] * v[i];
    }
    s1[threadIdx.x] = sum; s2[threadIdx.x] = sq;
    __syncthreads();
    for (int o = 128; o > 0; o >>= 1) {
        if ((int)threadIdx.x < o) {
            s1[threadIdx.x] += s1[threadIdx.x + o];
            s2[threadIdx.x] += s2[threadIdx.x + o];
        }
        __syncthreads();
    }
    const float mu  = s1[0] * (1.f / 1024.f);
    const float var = s2[0] * (1.f / 1024.f) - mu * mu;
    const float rs  = rsqrtf(var + 1e-5f);
#pragma unroll
    for (int i = 0; i < 4; i++) {
        const int c = threadIdx.x + i * 256;
        out[(size_t)row * 1024 + c] = f2bf((v[i] - mu) * rs * g[c] + b[c]);
    }
}

// ---------------------------------------------------------------------------
// GEMM: C[4096 x N] = A_bf16[4096 x K] @ W_f32[K x N]  (+ epilogue)
//  EPI 0: store bf16                        (QKV projection)
//  EPI 1: + bias + resid -> fp32 store      (out-proj / MLP2, residual update)
//  EPI 2: + bias, exact GELU -> bf16 store  (MLP1)
// Block tile 128x128, K-step 64, 8 waves = 4(M) x 2(N), wave tile 32x64.
// A tile staged via async global->LDS (ASYNCcnt); W converted fp32->bf16
// with packed cvt while transposing into LDS [n][k].
// ---------------------------------------------------------------------------
template <int EPI>
__global__ __launch_bounds__(256)
void gemm_bf16(const ush* __restrict__ A, const float* __restrict__ W,
               const float* __restrict__ bias, const float* __restrict__ resid,
               void* __restrict__ Cout, int N, int K)
{
    __shared__ ush As[128][72];   // [m][k], pad 8 (row stride 144B, 16B aligned)
    __shared__ ush Bs[128][72];   // [n][k] (transposed weight tile), bf16

    const int tid  = threadIdx.x;
    const int lane = tid & 31;
    const int wave = tid >> 5;
    const int wm   = wave & 3;          // 0..3 : 32-row group
    const int wn   = wave >> 2;         // 0..1 : 64-col group
    const int half = lane >> 4;         // 0/1
    const int l16  = lane & 15;
    const int blkM = blockIdx.y * 128;
    const int blkN = blockIdx.x * 128;

    v8f acc[2][4];
#pragma unroll
    for (int i = 0; i < 2; i++)
#pragma unroll
        for (int j = 0; j < 4; j++) { v8f z = ZERO8; acc[i][j] = z; }

    const int arow = tid >> 1;            // 0..127
    const int acol = (tid & 1) * 32;      // 0 or 32
    const int kp   = tid >> 3;            // 0..31 : k-pair index
    const int nb   = (tid & 7) * 16;      // 0..112

    const unsigned ldsA = (unsigned)(uintptr_t)&As[arow][acol];

    for (int k0 = 0; k0 < K; k0 += 64) {
        // ---- stage A tile (bf16): async global -> LDS, 64B per thread ----
        {
            unsigned long long ga =
                (unsigned long long)(uintptr_t)(A + (size_t)(blkM + arow) * K + k0 + acol);
            asm volatile(
                "global_load_async_to_lds_b128 %0, %1, off\n\t"
                "global_load_async_to_lds_b128 %0, %1, off offset:16\n\t"
                "global_load_async_to_lds_b128 %0, %1, off offset:32\n\t"
                "global_load_async_to_lds_b128 %0, %1, off offset:48"
                :: "v"(ldsA), "v"(ga) : "memory");
        }
        // ---- stage W tile: fp32 [k][n] -> bf16 LDS [n][k], packed pairs ----
        {
            const float* g0 = W + (size_t)(k0 + 2 * kp) * N + blkN + nb;
            const float* g1 = g0 + N;
            if (k0 + 64 < K) __builtin_prefetch(g0 + (size_t)64 * N, 0, 0);
#pragma unroll
            for (int j = 0; j < 16; j++) {
                *(unsigned*)&Bs[nb + j][2 * kp] = pack2bf(g0[j], g1[j]);
            }
        }
        asm volatile("s_wait_asynccnt 0x0" ::: "memory");
        __syncthreads();

        // ---- compute: two 32-k sub-steps, 16 WMMAs ----
#pragma unroll
        for (int ks = 0; ks < 2; ks++) {
            Frag a[2], b[4];
#pragma unroll
            for (int mi = 0; mi < 2; mi++) {
                const ush* p = &As[wm * 32 + mi * 16 + l16][ks * 32];
                a[mi].q[0] = *(const uint4*)(p + half * 8);
                a[mi].q[1] = *(const uint4*)(p + 16 + half * 8);
            }
#pragma unroll
            for (int ni = 0; ni < 4; ni++) {
                const ush* p = &Bs[wn * 64 + ni * 16 + l16][ks * 32];
                b[ni].q[0] = *(const uint4*)(p + half * 8);
                b[ni].q[1] = *(const uint4*)(p + 16 + half * 8);
            }
#pragma unroll
            for (int mi = 0; mi < 2; mi++)
#pragma unroll
                for (int ni = 0; ni < 4; ni++)
                    acc[mi][ni] = wmma_bf16(a[mi], b[ni], acc[mi][ni]);
        }
        __syncthreads();
    }

    // epilogue: C layout -> row = base + i + 8*half, col = base + l16
#pragma unroll
    for (int mi = 0; mi < 2; mi++)
#pragma unroll
        for (int ni = 0; ni < 4; ni++)
#pragma unroll
            for (int i = 0; i < 8; i++) {
                const int row = blkM + wm * 32 + mi * 16 + i + half * 8;
                const int col = blkN + wn * 64 + ni * 16 + l16;
                const size_t idx = (size_t)row * N + col;
                float v = acc[mi][ni][i];
                if (EPI == 0) {
                    ((ush*)Cout)[idx] = f2bf(v);
                } else if (EPI == 1) {
                    ((float*)Cout)[idx] = v + bias[col] + resid[idx];
                } else {
                    float t = v + bias[col];
                    t = 0.5f * t * (1.f + erff(t * 0.70710678118f));
                    ((ush*)Cout)[idx] = f2bf(t);
                }
            }
}

// ---------------------------------------------------------------------------
// Transpose V slice: qkv[b, n, 2048 + h*64 + d] -> vT[(b*16+h)*64 + d][n]
// ---------------------------------------------------------------------------
__global__ __launch_bounds__(256)
void transpose_v(const ush* __restrict__ qkv, ush* __restrict__ vT)
{
    __shared__ ush t[64][72];
    const int b = blockIdx.z, h = blockIdx.y, n0 = blockIdx.x * 64;
    const int r  = threadIdx.x >> 2;        // 0..63
    const int db = (threadIdx.x & 3) * 16;  // 0..48
    const ush* src = qkv + (size_t)(b * 1024 + n0 + r) * 3072 + 2048 + h * 64 + db;
#pragma unroll
    for (int j = 0; j < 16; j++) t[r][db + j] = src[j];
    __syncthreads();
    ush* dst = vT + ((size_t)(b * 16 + h) * 64 + r) * 1024 + n0 + db;
#pragma unroll
    for (int j = 0; j < 16; j++) dst[j] = t[db + j][r];
}

// ---------------------------------------------------------------------------
// Flash-style attention. One wave = 16 query rows; 8 waves/block = 128 rows.
// grid = (8, HEADS, B). Online softmax over 32-key chunks; WMMA for QK^T and PV.
// P re-layout bounces through a wave-private LDS tile (no block barrier needed:
// same-wave DS ops are in-order; s_wait_dscnt + compiler fence for safety).
// ---------------------------------------------------------------------------
__global__ __launch_bounds__(256)
void attn_kernel(const ush* __restrict__ qkv, const ush* __restrict__ vT,
                 const int* __restrict__ mnp, const int* __restrict__ mbert,
                 ush* __restrict__ out)
{
    __shared__ ush pbuf[8][16][40];   // per-wave P bounce tile
    const int b = blockIdx.z, h = blockIdx.y;
    const int wave = threadIdx.x >> 5, lane = threadIdx.x & 31;
    const int half = lane >> 4, l16 = lane & 15;
    const int qbase = blockIdx.x * 128 + wave * 16;
    const float SCALE = 0.125f;       // 64^-0.5

    const ush* qptr  = qkv + (size_t)(b * 1024 + qbase) * 3072 + h * 64;
    const ush* kptr  = qkv + (size_t)(b * 1024) * 3072 + 1024 + h * 64;
    const ush* vtptr = vT + ((size_t)(b * 16 + h) * 64) * 1024;

    // Q A-fragments for d-chunks {0..31},{32..63}; A row = l16
    Frag qa[2];
    {
        const ush* p = qptr + (size_t)l16 * 3072;
#pragma unroll
        for (int dc = 0; dc < 2; dc++) {
            qa[dc].q[0] = *(const uint4*)(p + dc * 32 + half * 8);
            qa[dc].q[1] = *(const uint4*)(p + dc * 32 + 16 + half * 8);
        }
    }
    // row-side np mask for my 8 rows (row = qbase + i + 8*half)
    bool rok[8];
#pragma unroll
    for (int i = 0; i < 8; i++) rok[i] = mnp[b * 1024 + qbase + i + 8 * half] != 0;

    float m_i[8], l_i[8];
    v8f O[4];
#pragma unroll
    for (int i = 0; i < 8; i++) { m_i[i] = -1e30f; l_i[i] = 0.f; }
#pragma unroll
    for (int t = 0; t < 4; t++) { v8f z = ZERO8; O[t] = z; }

    for (int kc = 0; kc < 1024; kc += 32) {
        // scores for two 16-key tiles, reduce over d = 64 (2 WMMAs each)
        v8f s[2];
#pragma unroll
        for (int t = 0; t < 2; t++) { v8f z = ZERO8; s[t] = z; }
#pragma unroll
        for (int t = 0; t < 2; t++)
#pragma unroll
            for (int dc = 0; dc < 2; dc++) {
                Frag kb;   // B: col = key (l16), K = d (half*8..+7 and 16+half*8..+7 within chunk)
                const ush* p = kptr + (size_t)(kc + t * 16 + l16) * 3072 + dc * 32;
                kb.q[0] = *(const uint4*)(p + half * 8);
                kb.q[1] = *(const uint4*)(p + 16 + half * 8);
                s[t] = wmma_bf16(qa[dc], kb, s[t]);
            }
        // column masks for my two keys
        bool cok[2];
#pragma unroll
        for (int t = 0; t < 2; t++) {
            const int key = kc + t * 16 + l16;
            cok[t] = (mnp[b * 1024 + key] != 0) && (mbert[b * 1024 + key] == 0);
        }
        // scale, mask, online softmax update
        float p0[8], p1[8];
#pragma unroll
        for (int i = 0; i < 8; i++) {
            p0[i] = (rok[i] && cok[0]) ? s[0][i] * SCALE : -1000.f;
            p1[i] = (rok[i] && cok[1]) ? s[1][i] * SCALE : -1000.f;
        }
#pragma unroll
        for (int i = 0; i < 8; i++) {
            float mx = fmaxf(p0[i], p1[i]);
            for (int o = 1; o < 16; o <<= 1) mx = fmaxf(mx, __shfl_xor(mx, o, 32));
            const float mnew  = fmaxf(m_i[i], mx);
            const float alpha = __expf(m_i[i] - mnew);
            p0[i] = __expf(p0[i] - mnew);
            p1[i] = __expf(p1[i] - mnew);
            float rs = p0[i] + p1[i];
            for (int o = 1; o < 16; o <<= 1) rs += __shfl_xor(rs, o, 32);
            l_i[i] = l_i[i] * alpha + rs;
            m_i[i] = mnew;
#pragma unroll
            for (int t = 0; t < 4; t++) O[t][i] *= alpha;
        }
        // C-layout P -> wave-private LDS -> A-fragment (16 queries x 32 keys)
#pragma unroll
        for (int i = 0; i < 8; i++) {
            pbuf[wave][i + 8 * half][l16]      = f2bf(p0[i]);
            pbuf[wave][i + 8 * half][16 + l16] = f2bf(p1[i]);
        }
        asm volatile("s_wait_dscnt 0x0" ::: "memory");
        Frag pa;
        {
            const ush* p = &pbuf[wave][l16][0];
            pa.q[0] = *(const uint4*)(p + half * 8);
            pa.q[1] = *(const uint4*)(p + 16 + half * 8);
        }
        // O += P @ V  (4 d-tiles; B: col = d, K = key, contiguous in vT row)
#pragma unroll
        for (int t = 0; t < 4; t++) {
            Frag vb;
            const ush* p = vtptr + (size_t)(t * 16 + l16) * 1024 + kc;
            vb.q[0] = *(const uint4*)(p + half * 8);
            vb.q[1] = *(const uint4*)(p + 16 + half * 8);
            O[t] = wmma_bf16(pa, vb, O[t]);
        }
    }
    // finalize + store bf16
#pragma unroll
    for (int t = 0; t < 4; t++)
#pragma unroll
        for (int i = 0; i < 8; i++) {
            const int row = qbase + i + 8 * half;
            const int d   = t * 16 + l16;
            out[(size_t)(b * 1024 + row) * 1024 + h * 64 + d] = f2bf(O[t][i] / l_i[i]);
        }
}

// ---------------------------------------------------------------------------
extern "C" void kernel_launch(void* const* d_in, const int* in_sizes, int n_in,
                              void* d_out, int out_size, void* d_ws, size_t ws_size,
                              hipStream_t stream)
{
    (void)in_sizes; (void)n_in; (void)out_size; (void)ws_size;
    const float* x     = (const float*)d_in[0];
    const float* Wqkv  = (const float*)d_in[1];
    const float* Wout  = (const float*)d_in[2];
    const float* bout  = (const float*)d_in[3];
    const float* ln1g  = (const float*)d_in[4];
    const float* ln1b  = (const float*)d_in[5];
    const float* W1    = (const float*)d_in[6];
    const float* b1    = (const float*)d_in[7];
    const float* W2    = (const float*)d_in[8];
    const float* b2    = (const float*)d_in[9];
    const float* ln2g  = (const float*)d_in[10];
    const float* ln2b  = (const float*)d_in[11];
    const int*   mnp   = (const int*)d_in[12];
    const int*   mbert = (const int*)d_in[13];

    char* ws = (char*)d_ws;
    float* xr   = (float*)(ws);                        // 16 MiB residual stream
    ush*   h    = (ush*)(ws + (16u << 20));            //  8 MiB LN output (bf16)
    ush*   qkvb = (ush*)(ws + (24u << 20));            // 24 MiB qkv (bf16)
    ush*   vT   = (ush*)(ws + (48u << 20));            //  8 MiB V transposed
    ush*   aout = (ush*)(ws + (56u << 20));            //  8 MiB attention out
    ush*   mlph = (ush*)(ws + (64u << 20));            // 32 MiB MLP hidden

    hipMemcpyAsync(xr, x, (size_t)4096 * 1024 * 4, hipMemcpyDeviceToDevice, stream);

    for (int l = 0; l < 4; l++) {
        ln_kernel<<<4096, 256, 0, stream>>>(xr, ln1g + l * 1024, ln1b + l * 1024, h);
        gemm_bf16<0><<<dim3(3072 / 128, 32), 256, 0, stream>>>(
            h, Wqkv + (size_t)l * 1024 * 3072, nullptr, nullptr, qkvb, 3072, 1024);
        transpose_v<<<dim3(16, 16, 4), 256, 0, stream>>>(qkvb, vT);
        attn_kernel<<<dim3(8, 16, 4), 256, 0, stream>>>(qkvb, vT, mnp, mbert, aout);
        gemm_bf16<1><<<dim3(1024 / 128, 32), 256, 0, stream>>>(
            aout, Wout + (size_t)l * 1024 * 1024, bout + l * 1024, xr, xr, 1024, 1024);
        ln_kernel<<<4096, 256, 0, stream>>>(xr, ln2g + l * 1024, ln2b + l * 1024, h);
        gemm_bf16<2><<<dim3(4096 / 128, 32), 256, 0, stream>>>(
            h, W1 + (size_t)l * 1024 * 4096, b1 + l * 4096, nullptr, mlph, 4096, 1024);
        gemm_bf16<1><<<dim3(1024 / 128, 32), 256, 0, stream>>>(
            mlph, W2 + (size_t)l * 4096 * 1024, b2 + l * 1024, xr, xr, 1024, 4096);
    }

    hipMemcpyAsync(d_out, xr, (size_t)4096 * 1024 * 4, hipMemcpyDeviceToDevice, stream);
}
